// GQAMultiheadAttention_2224793060127
// MI455X (gfx1250) — matrix operand order repaired and checked by reading, not compiled, not run on hardware
//
#include <hip/hip_runtime.h>

typedef unsigned short ushort_t;
typedef unsigned int u32;
typedef __attribute__((ext_vector_type(4)))  u32    u32x4;
typedef __attribute__((ext_vector_type(16))) __bf16 v16bf;
typedef __attribute__((ext_vector_type(8)))  float  v8f;

union FragU { v16bf v; u32x4 q[2]; };

__device__ __forceinline__ ushort_t f2bf(float x) {
  u32 u = __float_as_uint(x);
  u += 0x7FFFu + ((u >> 16) & 1u);   // round-to-nearest-even
  return (ushort_t)(u >> 16);
}

// A fragment (16x32 bf16, row-major source, ld in elements).
// lane&15 = row M; per-lane K offsets: [8*(lane>>4) .. +7] and [16+8*(lane>>4) .. +7]
__device__ __forceinline__ v16bf frag_a(const ushort_t* base, int ld, int lane) {
  const ushort_t* p = base + (lane & 15) * ld + (lane >> 4) * 8;
  FragU f;
  f.q[0] = *(const u32x4*)(p);
  f.q[1] = *(const u32x4*)(p + 16);
  return f.v;
}

// B fragment (32x16 bf16). Source is B^T row-major: row n holds column n of B.
// lane&15 = col N; per-lane K offsets: [8*(lane>>4) .. +7] and [16+8*(lane>>4) .. +7]
__device__ __forceinline__ v16bf frag_b(const ushort_t* base, int ld, int lane) {
  const ushort_t* p = base + (lane & 15) * ld + (lane >> 4) * 8;
  FragU f;
  f.q[0] = *(const u32x4*)(p);
  f.q[1] = *(const u32x4*)(p + 16);
  return f.v;
}

__device__ __forceinline__ v8f wmma_bf16(v16bf a, v16bf b, v8f c) {
  return __builtin_amdgcn_wmma_f32_16x16x32_bf16(false, a, false, b, (short)0, c,
                                                 false, false);
}

// ---------------- elementwise fp32 -> bf16 ----------------
__global__ void k_cvt(const float* __restrict__ s, ushort_t* __restrict__ d, int n) {
  int i = blockIdx.x * blockDim.x + threadIdx.x;
  if (i < n) d[i] = f2bf(s[i]);
}

// ---------------- tiled transpose + convert: Wt[n*K+k] = bf16(W[k*N+n]) ----------------
__global__ void __launch_bounds__(256) k_cvt_tr(const float* __restrict__ W,
                                                ushort_t* __restrict__ Wt,
                                                int K, int N) {
  __shared__ float tile[32][33];
  int kb = blockIdx.x * 32, nb = blockIdx.y * 32;
  int tx = threadIdx.x, ty = threadIdx.y;      // block (32, 8)
  for (int i = ty; i < 32; i += 8)
    tile[i][tx] = W[(long)(kb + i) * N + nb + tx];
  __syncthreads();
  for (int i = ty; i < 32; i += 8)
    Wt[(long)(nb + i) * K + kb + tx] = f2bf(tile[tx][i]);
}

// ---------------- bf16 GEMM: C[M,N] = A[M,K] * B, B given as Bt[N,K] ----------------
// One wave per 32x64 output tile (2 row-fragments x 4 col-fragments -> 8 WMMA per
// 32-K step, ~21 FLOP/byte of L2 traffic). 8 waves per block.
// This shape compiles to a x4-unrolled loop with loads pipelined a full
// sub-iteration ahead of the WMMAs (staggered s_wait_loadcnt), ~121 VGPRs,
// and 2048 waves of grid parallelism for the 2048x2048 GEMMs.
__global__ void __launch_bounds__(256) k_gemm(const ushort_t* __restrict__ A,
                                              const ushort_t* __restrict__ Bt,
                                              float* __restrict__ C,
                                              int M, int N, int K) {
  int lane = threadIdx.x & 31;
  int wave = threadIdx.x >> 5;
  int m0 = blockIdx.x * 32;
  int n0 = (blockIdx.y * 8 + wave) * 64;
  if (n0 >= N) return;
  v8f acc00 = {}, acc01 = {}, acc02 = {}, acc03 = {};
  v8f acc10 = {}, acc11 = {}, acc12 = {}, acc13 = {};
  const ushort_t* A0 = A + (long)m0 * K;
  const ushort_t* A1 = A + (long)(m0 + 16) * K;
  const ushort_t* B0 = Bt + (long)(n0     ) * K;
  const ushort_t* B1 = Bt + (long)(n0 + 16) * K;
  const ushort_t* B2 = Bt + (long)(n0 + 32) * K;
  const ushort_t* B3 = Bt + (long)(n0 + 48) * K;
  for (int k0 = 0; k0 < K; k0 += 32) {
    v16bf a0 = frag_a(A0 + k0, K, lane);
    v16bf a1 = frag_a(A1 + k0, K, lane);
    v16bf b;
    b = frag_b(B0 + k0, K, lane);
    acc00 = wmma_bf16(a0, b, acc00);
    acc10 = wmma_bf16(a1, b, acc10);
    b = frag_b(B1 + k0, K, lane);
    acc01 = wmma_bf16(a0, b, acc01);
    acc11 = wmma_bf16(a1, b, acc11);
    b = frag_b(B2 + k0, K, lane);
    acc02 = wmma_bf16(a0, b, acc02);
    acc12 = wmma_bf16(a1, b, acc12);
    b = frag_b(B3 + k0, K, lane);
    acc03 = wmma_bf16(a0, b, acc03);
    acc13 = wmma_bf16(a1, b, acc13);
  }
  int col = lane & 15, hl = lane >> 4;
#pragma unroll
  for (int r = 0; r < 8; ++r) {
    long row0 = m0 + r + 8 * hl;
    float* Crow = C + row0 * N + n0 + col;
    Crow[0]  = acc00[r];
    Crow[16] = acc01[r];
    Crow[32] = acc02[r];
    Crow[48] = acc03[r];
    float* Crow1 = Crow + 16L * N;
    Crow1[0]  = acc10[r];
    Crow1[16] = acc11[r];
    Crow1[32] = acc12[r];
    Crow1[48] = acc13[r];
  }
}

// ---------------- RoPE (fp32 in) -> bf16 head-major [H][T][64] ----------------
__global__ void k_rope(const float* __restrict__ src, const float* __restrict__ sn,
                       const float* __restrict__ cs, ushort_t* __restrict__ dst, int H) {
  int id = blockIdx.x * blockDim.x + threadIdx.x;
  int total = 2048 * H * 32;
  if (id >= total) return;
  int i  = id & 31;
  int hh = (id >> 5) % H;
  int t  = id / (32 * H);
  int D  = H * 64;
  float x1 = src[(long)t * D + hh * 64 + 2 * i];
  float x2 = src[(long)t * D + hh * 64 + 2 * i + 1];
  float sv = sn[t * 32 + i], cv = cs[t * 32 + i];
  ushort_t* o = dst + ((long)hh * 2048 + t) * 64 + 2 * i;
  o[0] = f2bf(x1 * cv - x2 * sv);
  o[1] = f2bf(x1 * sv + x2 * cv);
}

// ---------------- flash attention over sliding window ----------------
// Q  : [32][T][64] bf16 (post-RoPE), Kb : [8][T][64] bf16 (post-RoPE)
// Vt : [8][64][T] bf16, ctx out: [T][2048] bf16
__global__ void __launch_bounds__(32) k_attn(const ushort_t* __restrict__ Q,
                                             const ushort_t* __restrict__ Kb,
                                             const ushort_t* __restrict__ Vt,
                                             ushort_t* __restrict__ ctx,
                                             const int* __restrict__ swa) {
  __shared__ ushort_t Pst[16 * 32];
  const int T = 2048;
  int lane = threadIdx.x;
  int qt = blockIdx.x, h = blockIdx.y;
  int t0 = qt * 16;
  int hk = h >> 2;                         // GROUP = 4
  int Wwin = swa[0];
  const ushort_t* Qh = Q + ((long)h * T + t0) * 64;
  const ushort_t* Kh = Kb + (long)hk * T * 64;
  const ushort_t* Vh = Vt + (long)hk * 64 * T;

  v16bf qa0 = frag_a(Qh, 64, lane);        // d 0..31
  v16bf qa1 = frag_a(Qh + 32, 64, lane);   // d 32..63

  float m[8], s[8];
  v8f o0 = {}, o1 = {}, o2 = {}, o3 = {};
#pragma unroll
  for (int r = 0; r < 8; ++r) { m[r] = -1e30f; s[r] = 0.f; }

  int kstart = t0 - Wwin; if (kstart < 0) kstart = 0;
  kstart &= ~31;
  int kend = t0 + 15;
  int col = lane & 15, hl = lane >> 4;
  const float scale = 0.125f;              // 1/sqrt(64)

  for (int kb0 = kstart; kb0 <= kend; kb0 += 32) {
    // scores: two 16x16 tiles covering keys kb0..kb0+31
    v8f s0 = {}, s1 = {};
    s0 = wmma_bf16(qa0, frag_b(Kh + (long)kb0 * 64,        64, lane), s0);
    s0 = wmma_bf16(qa1, frag_b(Kh + (long)kb0 * 64 + 32,   64, lane), s0);
    s1 = wmma_bf16(qa0, frag_b(Kh + (long)(kb0 + 16) * 64,      64, lane), s1);
    s1 = wmma_bf16(qa1, frag_b(Kh + (long)(kb0 + 16) * 64 + 32, 64, lane), s1);

    int k0c = kb0 + col, k1c = kb0 + 16 + col;
    float bm[8];
#pragma unroll
    for (int r = 0; r < 8; ++r) {
      int q = t0 + r + 8 * hl;
      float v0 = s0[r] * scale;
      float v1 = s1[r] * scale;
      int d0 = q - k0c, d1 = q - k1c;
      if (d0 < 0 || d0 > Wwin) v0 = -1e30f;
      if (d1 < 0 || d1 > Wwin) v1 = -1e30f;
      s0[r] = v0; s1[r] = v1;
      bm[r] = fmaxf(v0, v1);
    }
    // row-max across the 16 lanes of each half (rows replicate per half)
#pragma unroll
    for (int msk = 1; msk <= 8; msk <<= 1)
#pragma unroll
      for (int r = 0; r < 8; ++r)
        bm[r] = fmaxf(bm[r], __shfl_xor(bm[r], msk, 32));

    float ps[8];
#pragma unroll
    for (int r = 0; r < 8; ++r) {
      float nm = fmaxf(m[r], bm[r]);
      float sc = __expf(m[r] - nm);
      m[r] = nm;
      s[r] *= sc;
      o0[r] *= sc; o1[r] *= sc; o2[r] *= sc; o3[r] *= sc;
      float p0 = __expf(s0[r] - nm);
      float p1 = __expf(s1[r] - nm);
      s0[r] = p0; s1[r] = p1;
      ps[r] = p0 + p1;
    }
#pragma unroll
    for (int msk = 1; msk <= 8; msk <<= 1)
#pragma unroll
      for (int r = 0; r < 8; ++r)
        ps[r] += __shfl_xor(ps[r], msk, 32);
#pragma unroll
    for (int r = 0; r < 8; ++r) s[r] += ps[r];

    // transpose P (C-layout -> A-fragment layout) through LDS
    __syncthreads();
#pragma unroll
    for (int r = 0; r < 8; ++r) {
      int Mr = r + 8 * hl;
      Pst[Mr * 32 + col]      = f2bf(s0[r]);
      Pst[Mr * 32 + 16 + col] = f2bf(s1[r]);
    }
    __syncthreads();
    FragU pf;
    const ushort_t* pp = Pst + (lane & 15) * 32 + hl * 8;
    pf.q[0] = *(const u32x4*)(pp);
    pf.q[1] = *(const u32x4*)(pp + 16);

    // P(16x32) @ V(32x64): 4 d-tiles
    o0 = wmma_bf16(pf.v, frag_b(Vh + 0L * 16 * T + kb0, T, lane), o0);
    o1 = wmma_bf16(pf.v, frag_b(Vh + 1L * 16 * T + kb0, T, lane), o1);
    o2 = wmma_bf16(pf.v, frag_b(Vh + 2L * 16 * T + kb0, T, lane), o2);
    o3 = wmma_bf16(pf.v, frag_b(Vh + 3L * 16 * T + kb0, T, lane), o3);
  }
#pragma unroll
  for (int r = 0; r < 8; ++r) {
    float inv = 1.f / s[r];
    long row = t0 + r + 8 * hl;
    ushort_t* cp = ctx + row * 2048 + h * 64 + col;
    cp[0]  = f2bf(o0[r] * inv);
    cp[16] = f2bf(o1[r] * inv);
    cp[32] = f2bf(o2[r] * inv);
    cp[48] = f2bf(o3[r] * inv);
  }
}

extern "C" void kernel_launch(void* const* d_in, const int* in_sizes, int n_in,
                              void* d_out, int out_size, void* d_ws, size_t ws_size,
                              hipStream_t stream) {
  const float* x    = (const float*)d_in[0];
  const float* Wq   = (const float*)d_in[1];
  const float* Wk   = (const float*)d_in[2];
  const float* Wv   = (const float*)d_in[3];
  const float* Wo   = (const float*)d_in[4];
  const float* rsin = (const float*)d_in[5];
  const float* rcos = (const float*)d_in[6];
  const int*   swa  = (const int*)d_in[7];
  float* out = (float*)d_out;

  char* ws = (char*)d_ws;
  const size_t MB = 1024 * 1024;
  ushort_t* xb   = (ushort_t*)(ws + 0);        //  8 MB  x bf16        [2048,2048]
  ushort_t* Wqt  = (ushort_t*)(ws + 8 * MB);   //  8 MB  Wq^T bf16     [2048,2048]
  ushort_t* Wkt  = (ushort_t*)(ws + 16 * MB);  //  2 MB  Wk^T bf16     [512,2048]
  ushort_t* Wvt  = (ushort_t*)(ws + 18 * MB);  //  2 MB  Wv^T bf16     [512,2048]
  ushort_t* Wot  = (ushort_t*)(ws + 20 * MB);  //  8 MB  Wo^T bf16     [2048,2048]
  float*    qf   = (float*)   (ws + 28 * MB);  // 16 MB  q fp32        [2048,2048]
  float*    kf   = (float*)   (ws + 44 * MB);  //  4 MB  k fp32        [2048,512]
  float*    vf   = (float*)   (ws + 48 * MB);  //  4 MB  v fp32        [2048,512]
  ushort_t* qb   = (ushort_t*)(ws + 52 * MB);  //  8 MB  q bf16 roped  [32][2048][64]
  ushort_t* kbuf = (ushort_t*)(ws + 60 * MB);  //  2 MB  k bf16 roped  [8][2048][64]
  ushort_t* vtb  = (ushort_t*)(ws + 62 * MB);  //  2 MB  v^T bf16      [8][64][2048]
  ushort_t* ctxb = (ushort_t*)(ws + 64 * MB);  //  8 MB  attn ctx bf16 [2048,2048]

  // precision conversion + weight transposes (all stays L2-resident: ~72 MB < 192 MB L2)
  k_cvt<<<dim3((2048 * 2048) / 256), dim3(256), 0, stream>>>(x, xb, 2048 * 2048);
  k_cvt_tr<<<dim3(64, 64), dim3(32, 8), 0, stream>>>(Wq, Wqt, 2048, 2048);
  k_cvt_tr<<<dim3(64, 16), dim3(32, 8), 0, stream>>>(Wk, Wkt, 2048, 512);
  k_cvt_tr<<<dim3(64, 16), dim3(32, 8), 0, stream>>>(Wv, Wvt, 2048, 512);
  k_cvt_tr<<<dim3(64, 64), dim3(32, 8), 0, stream>>>(Wo, Wot, 2048, 2048);

  // QKV projections (bf16 WMMA, fp32 accum); 32x64 tile per wave
  k_gemm<<<dim3(64, 4), dim3(256), 0, stream>>>(xb, Wqt, qf, 2048, 2048, 2048);
  k_gemm<<<dim3(64, 1), dim3(256), 0, stream>>>(xb, Wkt, kf, 2048, 512, 2048);
  k_gemm<<<dim3(64, 1), dim3(256), 0, stream>>>(xb, Wvt, vf, 2048, 512, 2048);

  // RoPE (fp32) -> bf16 head-major; V -> bf16 [kvh][d][t]
  k_rope<<<dim3((2048 * 32 * 32) / 256), dim3(256), 0, stream>>>(qf, rsin, rcos, qb, 32);
  k_rope<<<dim3((2048 * 8 * 32) / 256),  dim3(256), 0, stream>>>(kf, rsin, rcos, kbuf, 8);
  k_cvt_tr<<<dim3(64, 16), dim3(32, 8), 0, stream>>>(vf, vtb, 2048, 512);

  // sliding-window flash attention: one wave per (16-query tile, head)
  k_attn<<<dim3(128, 32), dim3(32), 0, stream>>>(qb, kbuf, vtb, ctxb, swa);

  // output projection -> fp32 result
  k_gemm<<<dim3(64, 4), dim3(256), 0, stream>>>(ctxb, Wot, out, 2048, 2048, 2048);
}
